// CFModel_59734405152888
// MI455X (gfx1250) — matrix operand order repaired and checked by reading, not compiled, not run on hardware
//
#include <hip/hip_runtime.h>

// ---- problem constants (match reference) ----
#define NN 50000
#define EE 800000
#define RR 8
#define DD 128
#define LL 2
#define NEG_SLOPE 0.01f

typedef __attribute__((ext_vector_type(16))) __bf16 v16bf;
typedef __attribute__((ext_vector_type(8)))  float  v8f;
typedef __attribute__((ext_vector_type(4)))  __bf16 bf16x4;

union AB16 { v16bf v; uint4 q[2]; };

// monotonic float<->uint mapping so float max can use global_atomic_max_u32
__device__ __forceinline__ unsigned f2mono(float f){
  unsigned u = __float_as_uint(f);
  return (u & 0x80000000u) ? ~u : (u | 0x80000000u);
}
__device__ __forceinline__ float mono2f(unsigned m){
  return (m & 0x80000000u) ? __uint_as_float(m ^ 0x80000000u) : __uint_as_float(~m);
}
__device__ __forceinline__ float lrelu(float v){ return v > 0.f ? v : NEG_SLOPE * v; }

// ---------------- utility kernels ----------------
__global__ void k_zero(unsigned* __restrict__ p, long n){
  long i = (long)blockIdx.x * blockDim.x + threadIdx.x;
  if (i < n) p[i] = 0u;
}

__global__ void k_cvt_bf16(const float* __restrict__ s, __bf16* __restrict__ d, long n){
  long i = (long)blockIdx.x * blockDim.x + threadIdx.x;
  if (i < n) d[i] = (__bf16)s[i];
}

// relation_W [r][d][e] f32  ->  Wt [r][e][d] bf16  (B operand wants K contiguous)
__global__ void k_transpose_relW(const float* __restrict__ W, __bf16* __restrict__ Wt){
  int i = blockIdx.x * 256 + threadIdx.x;
  if (i >= RR*DD*DD) return;
  int r = i / (DD*DD); int rem = i % (DD*DD);
  int e = rem / DD;    int d = rem % DD;
  Wt[i] = (__bf16)W[(size_t)r*DD*DD + (size_t)d*DD + e];
}

// h = entity_table[node_ids]; also bf16 copy; also output columns [0:128)
__global__ void k_gather(const int* __restrict__ nid, const float* __restrict__ etab,
                         float* __restrict__ h, __bf16* __restrict__ hb,
                         float* __restrict__ outp){
  int i = blockIdx.x * 256 + threadIdx.x;
  if (i >= NN*DD) return;
  int n = i >> 7, d = i & (DD-1);
  float v = etab[(size_t)nid[n]*DD + d];
  h[i]  = v;
  hb[i] = (__bf16)v;
  outp[(size_t)n*(3*DD) + d] = v;
}

// ---------------- proj GEMM: proj[n][r][e] = sum_d h[n,d] * W[r,d,e] ----------------
// grid (ceil(N/128), R), block 256 = 8 waves; each wave: 16 rows x 128 cols, K-loop 4x32.
__global__ void __launch_bounds__(256) k_proj(const __bf16* __restrict__ hb,
                                              const __bf16* __restrict__ Wt,
                                              __bf16* __restrict__ projb){
  __shared__ __align__(16) __bf16 Bs[DD*DD];   // 32 KB: Wt[r] = [e][d], K-contiguous
  const int r = blockIdx.y;
  { const uint4* s4 = (const uint4*)(Wt + (size_t)r*DD*DD);
    uint4* d4 = (uint4*)Bs;
    for (int i = threadIdx.x; i < DD*DD/8; i += 256) d4[i] = s4[i]; }
  __syncthreads();

  const int wave = threadIdx.x >> 5, lane = threadIdx.x & 31;
  const int m0 = blockIdx.x * 128 + wave * 16;
  if (m0 >= NN) return;
  const int hi = lane >> 4;                 // half-wave select
  int mrow = m0 + (lane & 15); if (mrow > NN-1) mrow = NN-1;
  const __bf16* pa = hb + (size_t)mrow*DD + hi*8;   // A-fragment lane base
  const int bcol = lane & 15;

  v8f acc[8];
  const v8f z = {0.f,0.f,0.f,0.f,0.f,0.f,0.f,0.f};
  #pragma unroll
  for (int t = 0; t < 8; ++t) acc[t] = z;

  #pragma unroll
  for (int k0 = 0; k0 < DD; k0 += 32){
    AB16 A;
    A.q[0] = *(const uint4*)(pa + k0);        // K = k0   + hi*8 .. +7
    A.q[1] = *(const uint4*)(pa + k0 + 16);   // K = k0+16+ hi*8 .. +7
    if (k0 + 32 < DD) __builtin_prefetch(pa + k0 + 32, 0, 1);
    #pragma unroll
    for (int t = 0; t < 8; ++t){
      AB16 B;
      const __bf16* pb = Bs + (t*16 + bcol)*DD + k0 + hi*8;
      B.q[0] = *(const uint4*)pb;             // K = k0   + hi*8 .. +7
      B.q[1] = *(const uint4*)(pb + 16);      // K = k0+16+ hi*8 .. +7
      acc[t] = __builtin_amdgcn_wmma_f32_16x16x32_bf16(
          false, A.v, false, B.v, (short)0, acc[t], false, false);
    }
  }

  const int rowadd = hi * 8;
  #pragma unroll
  for (int t = 0; t < 8; ++t){
    #pragma unroll
    for (int v = 0; v < 8; ++v){
      int row = m0 + rowadd + v;
      if (row < NN)
        projb[((size_t)row*RR + r)*DD + t*16 + bcol] = (__bf16)acc[t][v];
    }
  }
}

// ---------------- edge attention: att[e] = sum_d t_r[d] * tanh(h_r[d] + efeat[d]) ----------------
__global__ void __launch_bounds__(256) k_att(const __bf16* __restrict__ projb,
                                             const float* __restrict__ reltab,
                                             const int* __restrict__ rel,
                                             const int* __restrict__ src,
                                             const int* __restrict__ dst,
                                             float* __restrict__ att,
                                             unsigned* __restrict__ segmax){
  __shared__ float efs[RR*DD];                  // 4 KB relation table
  for (int i = threadIdx.x; i < RR*DD; i += 256) efs[i] = reltab[i];
  __syncthreads();
  const int wave = threadIdx.x >> 5, lane = threadIdx.x & 31;
  const long e = (long)blockIdx.x * 8 + wave;
  if (e >= EE) return;
  const int r = rel[e], s = src[e], d = dst[e];
  bf16x4 t4 = *(const bf16x4*)(projb + ((size_t)s*RR + r)*DD + lane*4);
  bf16x4 h4 = *(const bf16x4*)(projb + ((size_t)d*RR + r)*DD + lane*4);
  const float* pe = efs + r*DD + lane*4;
  float acc = 0.f;
  #pragma unroll
  for (int j = 0; j < 4; ++j)
    acc += (float)t4[j] * tanhf((float)h4[j] + pe[j]);
  #pragma unroll
  for (int off = 16; off > 0; off >>= 1) acc += __shfl_xor(acc, off, 32);
  if (lane == 0){ att[e] = acc; atomicMax(segmax + d, f2mono(acc)); }
}

__global__ void k_exp(const float* __restrict__ att, const unsigned* __restrict__ segmax,
                      const int* __restrict__ dst, float* __restrict__ ex,
                      float* __restrict__ segsum){
  long e = (long)blockIdx.x * 256 + threadIdx.x;
  if (e >= EE) return;
  int d = dst[e];
  float v = __expf(att[e] - mono2f(segmax[d]));
  ex[e] = v;
  unsafeAtomicAdd(segsum + d, v);
}

// h_nb[dst,:] += (ex/segsum[dst]) * h[src,:]
__global__ void __launch_bounds__(256) k_scatter(const float* __restrict__ ex,
                                                 const float* __restrict__ segsum,
                                                 const int* __restrict__ src,
                                                 const int* __restrict__ dst,
                                                 const float* __restrict__ h,
                                                 float* __restrict__ hnb){
  const int wave = threadIdx.x >> 5, lane = threadIdx.x & 31;
  const long e = (long)blockIdx.x * 8 + wave;
  if (e >= EE) return;
  const int s = src[e], d = dst[e];
  const float a = ex[e] / segsum[d];
  const float4 v = *(const float4*)(h + (size_t)s*DD + lane*4);
  float* p = hnb + (size_t)d*DD + lane*4;
  unsafeAtomicAdd(p + 0, a * v.x);
  unsafeAtomicAdd(p + 1, a * v.y);
  unsafeAtomicAdd(p + 2, a * v.z);
  unsafeAtomicAdd(p + 3, a * v.w);
}

// x1 = bf16(h + h_nb), x2 = bf16(h * h_nb), plus bf16(h_nb) for next layer's proj A
__global__ void k_prepx(const float* __restrict__ h, const float* __restrict__ hnb,
                        __bf16* __restrict__ x1b, __bf16* __restrict__ x2b,
                        __bf16* __restrict__ hbn){
  int i = blockIdx.x * 256 + threadIdx.x;
  if (i >= NN*DD) return;
  float a = h[i], b = hnb[i];
  x1b[i] = (__bf16)(a + b);
  x2b[i] = (__bf16)(a * b);
  hbn[i] = (__bf16)b;
}

// ---------------- residual output: out = lrelu(x1 @ W1^T) + lrelu(x2 @ W2^T) ----------------
// W row-major [e][d] is already K-contiguous for the B operand.
__global__ void __launch_bounds__(256) k_out(const __bf16* __restrict__ x1b,
                                             const __bf16* __restrict__ x2b,
                                             const __bf16* __restrict__ W1b,
                                             const __bf16* __restrict__ W2b,
                                             float* __restrict__ outp, int colbase){
  __shared__ __align__(16) __bf16 B1s[DD*DD];   // 32 KB
  __shared__ __align__(16) __bf16 B2s[DD*DD];   // 32 KB
  { const uint4* s1 = (const uint4*)W1b; const uint4* s2 = (const uint4*)W2b;
    uint4* d1 = (uint4*)B1s; uint4* d2 = (uint4*)B2s;
    for (int i = threadIdx.x; i < DD*DD/8; i += 256){ d1[i] = s1[i]; d2[i] = s2[i]; } }
  __syncthreads();

  const int wave = threadIdx.x >> 5, lane = threadIdx.x & 31;
  const int m0 = blockIdx.x * 128 + wave * 16;
  if (m0 >= NN) return;
  const int hi = lane >> 4;
  int mrow = m0 + (lane & 15); if (mrow > NN-1) mrow = NN-1;
  const __bf16* pa1 = x1b + (size_t)mrow*DD + hi*8;
  const __bf16* pa2 = x2b + (size_t)mrow*DD + hi*8;
  const int bcol = lane & 15;

  v8f acc1[8], acc2[8];
  const v8f z = {0.f,0.f,0.f,0.f,0.f,0.f,0.f,0.f};
  #pragma unroll
  for (int t = 0; t < 8; ++t){ acc1[t] = z; acc2[t] = z; }

  #pragma unroll
  for (int k0 = 0; k0 < DD; k0 += 32){
    AB16 A1, A2;
    A1.q[0] = *(const uint4*)(pa1 + k0);
    A1.q[1] = *(const uint4*)(pa1 + k0 + 16);
    A2.q[0] = *(const uint4*)(pa2 + k0);
    A2.q[1] = *(const uint4*)(pa2 + k0 + 16);
    #pragma unroll
    for (int t = 0; t < 8; ++t){
      AB16 B1, B2;
      const int bofs = (t*16 + bcol)*DD + k0 + hi*8;
      B1.q[0] = *(const uint4*)(B1s + bofs);
      B1.q[1] = *(const uint4*)(B1s + bofs + 16);
      B2.q[0] = *(const uint4*)(B2s + bofs);
      B2.q[1] = *(const uint4*)(B2s + bofs + 16);
      acc1[t] = __builtin_amdgcn_wmma_f32_16x16x32_bf16(
          false, A1.v, false, B1.v, (short)0, acc1[t], false, false);
      acc2[t] = __builtin_amdgcn_wmma_f32_16x16x32_bf16(
          false, A2.v, false, B2.v, (short)0, acc2[t], false, false);
    }
  }

  const int rowadd = hi * 8;
  #pragma unroll
  for (int t = 0; t < 8; ++t){
    #pragma unroll
    for (int v = 0; v < 8; ++v){
      int row = m0 + rowadd + v;
      if (row < NN){
        float o = lrelu(acc1[t][v]) + lrelu(acc2[t][v]);
        outp[(size_t)row*(3*DD) + colbase + t*16 + bcol] = o;
      }
    }
  }
}

// ---------------- host-side launch ----------------
extern "C" void kernel_launch(void* const* d_in, const int* in_sizes, int n_in,
                              void* d_out, int out_size, void* d_ws, size_t ws_size,
                              hipStream_t stream){
  const int*   node_ids = (const int*)  d_in[0];
  const int*   rel      = (const int*)  d_in[1];
  const int*   src      = (const int*)  d_in[2];
  const int*   dst      = (const int*)  d_in[3];
  const float* etab     = (const float*)d_in[4];
  const float* reltab   = (const float*)d_in[5];
  const float* relW     = (const float*)d_in[6];
  const float* fc1W     = (const float*)d_in[7];
  const float* fc2W     = (const float*)d_in[8];
  float* outp = (float*)d_out;
  (void)in_sizes; (void)n_in; (void)out_size; (void)ws_size;

  // bump allocator over d_ws (~200 MB total)
  char* w = (char*)d_ws; size_t off = 0;
  auto alloc = [&](size_t bytes)->void*{
    void* p = w + off; off = (off + bytes + 255) & ~(size_t)255; return p; };
  float*    hA     = (float*)   alloc((size_t)NN*DD*4);
  float*    hB     = (float*)   alloc((size_t)NN*DD*4);
  __bf16*   hb     = (__bf16*)  alloc((size_t)NN*DD*2);
  __bf16*   x1b    = (__bf16*)  alloc((size_t)NN*DD*2);
  __bf16*   x2b    = (__bf16*)  alloc((size_t)NN*DD*2);
  __bf16*   projb  = (__bf16*)  alloc((size_t)NN*RR*DD*2);
  __bf16*   Wt     = (__bf16*)  alloc((size_t)RR*DD*DD*2);
  __bf16*   W1b    = (__bf16*)  alloc((size_t)LL*DD*DD*2);
  __bf16*   W2b    = (__bf16*)  alloc((size_t)LL*DD*DD*2);
  float*    attb   = (float*)   alloc((size_t)EE*4);
  float*    exb    = (float*)   alloc((size_t)EE*4);
  unsigned* segmax = (unsigned*)alloc((size_t)NN*4);
  float*    segsum = (float*)   alloc((size_t)NN*4);

  // weight prep + embedding gather
  k_transpose_relW<<<(RR*DD*DD + 255)/256, 256, 0, stream>>>(relW, Wt);
  k_cvt_bf16<<<(LL*DD*DD + 255)/256, 256, 0, stream>>>(fc1W, W1b, (long)LL*DD*DD);
  k_cvt_bf16<<<(LL*DD*DD + 255)/256, 256, 0, stream>>>(fc2W, W2b, (long)LL*DD*DD);
  k_gather<<<(NN*DD + 255)/256, 256, 0, stream>>>(node_ids, etab, hA, hb, outp);

  float* hcur = hA; float* hnb = hB;
  const int MB = (NN + 127)/128;
  for (int l = 0; l < LL; ++l){
    k_zero<<<(NN + 255)/256, 256, 0, stream>>>(segmax, NN);
    k_zero<<<(NN + 255)/256, 256, 0, stream>>>((unsigned*)segsum, NN);
    k_zero<<<(int)(((long)NN*DD + 255)/256), 256, 0, stream>>>((unsigned*)hnb, (long)NN*DD);

    k_proj<<<dim3(MB, RR), 256, 0, stream>>>(hb, Wt, projb);
    k_att<<<(EE + 7)/8, 256, 0, stream>>>(projb, reltab, rel, src, dst, attb, segmax);
    k_exp<<<(EE + 255)/256, 256, 0, stream>>>(attb, segmax, dst, exb, segsum);
    k_scatter<<<(EE + 7)/8, 256, 0, stream>>>(exb, segsum, src, dst, hcur, hnb);
    k_prepx<<<(NN*DD + 255)/256, 256, 0, stream>>>(hcur, hnb, x1b, x2b, hb);
    k_out<<<MB, 256, 0, stream>>>(x1b, x2b, W1b + (size_t)l*DD*DD,
                                  W2b + (size_t)l*DD*DD, outp, (l + 1)*DD);
    float* tmp = hcur; hcur = hnb; hnb = tmp;
  }
}
